// RGCN_57999238365556
// MI455X (gfx1250) — matrix operand order repaired and checked by reading, not compiled, not run on hardware
//
#include <hip/hip_runtime.h>
#include <hip/hip_bf16.h>

// RGCN (3 layers, basis decomposition) + ConvTranspose2d decoder for MI455X (gfx1250).
// bf16 WMMA GEMM (self-loop folded in as a 9th relation block), async global->LDS
// staging of the packed weight fragments, fp32 atomic edge scatter, VALU decoder.

#define USE_ASYNC_LDS 1

typedef __attribute__((ext_vector_type(16))) __bf16 v16bf;
typedef __attribute__((ext_vector_type(8)))  float  v8f;

#define NNODES 115200
#define NEDGES 921600
#define RREL   8
#define NBASES 4
#define BGRAPH 128
#define GW     30
#define IW     90

// ---------------------------------------------------------------------------
// Build W_cat = [ sum_b comp[r,b]*bases[b]  (r=0..7) | loop_w ]  -> bf16, packed
// directly into the B-fragment layout used here (contiguous K, same as A):
//   fragment (kt, nt): 32 lanes x 16 bf16.
//   lane < 16 : column N = nt*16 + lane,    K = kt*32 + e        (e = 0..15)
//   lane >= 16: column N = nt*16 + lane-16, K = kt*32 + 16 + e
// ---------------------------------------------------------------------------
template<int K, int DOUT>
__global__ void pack_w_kernel(const float* __restrict__ bases,
                              const float* __restrict__ comp,
                              const float* __restrict__ loopw,
                              __bf16* __restrict__ bpack) {
    constexpr int KT    = K / 32;
    constexpr int NT    = ((RREL + 1) * DOUT) / 16;
    constexpr int TOTAL = KT * NT * 512;
    constexpr int NPROJ = RREL * DOUT;
    int idx = blockIdx.x * blockDim.x + threadIdx.x;
    if (idx >= TOTAL) return;
    int e    = idx & 15;
    int lane = (idx >> 4) & 31;
    int tile = idx >> 9;
    int nt   = tile % NT;
    int kt   = tile / NT;
    int k    = kt * 32 + ((lane >> 4) << 4) + e;
    int ncol = nt * 16 + (lane & 15);
    float val;
    if (ncol < NPROJ) {
        int r = ncol / DOUT;
        int j = ncol % DOUT;
        val = 0.f;
#pragma unroll
        for (int b = 0; b < NBASES; ++b)
            val += comp[r * NBASES + b] * bases[((size_t)b * K + k) * DOUT + j];
    } else {
        val = loopw[(size_t)k * DOUT + (ncol - NPROJ)];
    }
    bpack[idx] = (__bf16)val;
}

// ---------------------------------------------------------------------------
// GEMM:  [proj | agg] = A[N x K] (fp32 -> bf16) @ Wcat[K x 9*DOUT] (bf16)
// Block = 256 threads = 8 waves, each wave owns one 16-row strip.
// Wcat fragments staged once per block into LDS via async DMA, reused by all waves.
// ---------------------------------------------------------------------------
template<int K, int DOUT>
__global__ __launch_bounds__(256)
void gemm_wmma_kernel(const float* __restrict__ A,
                      const __bf16* __restrict__ bpack,
                      const float* __restrict__ bias,
                      float* __restrict__ proj,
                      float* __restrict__ agg) {
    constexpr int KT      = K / 32;
    constexpr int NT      = ((RREL + 1) * DOUT) / 16;
    constexpr int NPROJ   = RREL * DOUT;
    constexpr int NT_PROJ = NPROJ / 16;     // relation-block tiles
    constexpr int NT_AGG  = DOUT / 16;      // self-loop tiles
    constexpr int LDSE    = KT * NT * 512;  // bf16 elements (<= 36864 = 72KB)
    __shared__ __attribute__((aligned(32))) __bf16 ldsB[LDSE];

#if USE_ASYNC_LDS
    // CDNA5 async global->LDS DMA (GLOBAL_LOAD_ASYNC_TO_LDS_B128, ASYNCcnt)
    {
        const char* g = (const char*)bpack;
        for (int i = threadIdx.x; i < LDSE / 8; i += 256) {
            int lds_off = (int)(unsigned)(uintptr_t)&(((uint4*)ldsB)[i]);
            asm volatile("global_load_async_to_lds_b128 %0, %1, off"
                         :: "v"(lds_off), "v"(g + 16 * (size_t)i) : "memory");
        }
        asm volatile("s_wait_asynccnt 0x0" ::: "memory");
    }
#else
    {
        const uint4* s = (const uint4*)bpack;
        uint4*       d = (uint4*)ldsB;
        for (int i = threadIdx.x; i < LDSE / 8; i += 256) d[i] = s[i];
    }
#endif
    __syncthreads();

    const int lane    = threadIdx.x & 31;
    const int wave    = threadIdx.x >> 5;
    const int baseRow = (blockIdx.x * 8 + wave) * 16;
    const int mrow    = baseRow + (lane & 15);
    const int khalf   = lane >> 4;                  // 0 or 1

    // A fragments: contiguous layout -> lane<16: K {0..15}; lane>=16: K {16..31}
    v16bf afrag[KT];
#pragma unroll
    for (int kt = 0; kt < KT; ++kt) {
        const float* ap = A + (size_t)mrow * K + kt * 32 + khalf * 16;
        v16bf a;
#pragma unroll
        for (int e = 0; e < 8; ++e) a[e] = (__bf16)ap[e];
#pragma unroll
        for (int e = 0; e < 8; ++e) a[8 + e] = (__bf16)ap[8 + e];
        afrag[kt] = a;
    }

    const int ncol = lane & 15;
    const int rtop = baseRow + khalf * 8;

    // ---- relation blocks -> proj (compile-time loop, no branches) ----
    float* projp = proj + (size_t)rtop * NPROJ + ncol;
#pragma unroll
    for (int nt = 0; nt < NT_PROJ; ++nt) {
        v8f c = {0.f, 0.f, 0.f, 0.f, 0.f, 0.f, 0.f, 0.f};
#pragma unroll
        for (int kt = 0; kt < KT; ++kt) {
            v16bf b = *(const v16bf*)&ldsB[((kt * NT + nt) * 32 + lane) * 16];
            c = __builtin_amdgcn_wmma_f32_16x16x32_bf16(
                    false, afrag[kt], false, b, (short)0, c, false, false);
        }
#pragma unroll
        for (int v = 0; v < 8; ++v) projp[(size_t)v * NPROJ + nt * 16] = c[v];
    }

    // ---- self-loop block (+bias) -> agg ----
    float bj[NT_AGG];
#pragma unroll
    for (int t = 0; t < NT_AGG; ++t) bj[t] = bias[t * 16 + ncol];

    float* aggp = agg + (size_t)rtop * DOUT + ncol;
#pragma unroll
    for (int t = 0; t < NT_AGG; ++t) {
        v8f c = {0.f, 0.f, 0.f, 0.f, 0.f, 0.f, 0.f, 0.f};
#pragma unroll
        for (int kt = 0; kt < KT; ++kt) {
            v16bf b = *(const v16bf*)&ldsB[((kt * NT + NT_PROJ + t) * 32 + lane) * 16];
            c = __builtin_amdgcn_wmma_f32_16x16x32_bf16(
                    false, afrag[kt], false, b, (short)0, c, false, false);
        }
#pragma unroll
        for (int v = 0; v < 8; ++v) aggp[(size_t)v * DOUT + t * 16] = c[v] + bj[t];
    }
}

// ---------------------------------------------------------------------------
// Edge scatter: agg[dst] += proj[src, etype] * norm
// One thread per (edge, 4 channels): float4 gather + 4 fp32 atomics.
// ---------------------------------------------------------------------------
template<int DOUT>
__global__ void scatter_kernel(const float* __restrict__ proj,
                               const int* __restrict__ src,
                               const int* __restrict__ dst,
                               const int* __restrict__ ety,
                               const float* __restrict__ norm,
                               float* __restrict__ agg) {
    constexpr int NPROJ = RREL * DOUT;
    constexpr int VJ    = DOUT / 4;
    int idx = blockIdx.x * blockDim.x + threadIdx.x;
    if (idx >= NEDGES * VJ) return;
    int e = idx / VJ;
    int q = idx - e * VJ;
    const float4 p = *(const float4*)(proj + (size_t)src[e] * NPROJ + ety[e] * DOUT + q * 4);
    float nv = norm[e];
    float* a = agg + (size_t)dst[e] * DOUT + q * 4;
    atomicAdd(a + 0, p.x * nv);
    atomicAdd(a + 1, p.y * nv);
    atomicAdd(a + 2, p.z * nv);
    atomicAdd(a + 3, p.w * nv);
}

// ---------------------------------------------------------------------------
// Elementwise activation: leaky-relu (mode 0) or sigmoid (mode 1)
// ---------------------------------------------------------------------------
__global__ void act_kernel(const float* __restrict__ in, float* __restrict__ out,
                           int n, int sigmoid_mode) {
    int i = blockIdx.x * blockDim.x + threadIdx.x;
    if (i >= n) return;
    float x = in[i];
    out[i] = sigmoid_mode ? (1.f / (1.f + __expf(-x)))
                          : (x >= 0.f ? x : 0.01f * x);
}

// ---------------------------------------------------------------------------
// ConvTranspose2d(32->1, k=7, s=3, p=2) gather form + sigmoid.
// x: (B, 30, 30, 32) node-major NHWC (== logits rows). w: (32,1,7,7) torch layout.
// ---------------------------------------------------------------------------
__global__ void convt_sigmoid_kernel(const float* __restrict__ x,
                                     const float* __restrict__ w,
                                     const float* __restrict__ bconv,
                                     float* __restrict__ out) {
    int idx = blockIdx.x * blockDim.x + threadIdx.x;
    const int total = BGRAPH * IW * IW;
    if (idx >= total) return;
    int ox = idx % IW;
    int t  = idx / IW;
    int oy = t % IW;
    int b  = t / IW;
    float acc = bconv[0];
#pragma unroll
    for (int kh = 0; kh < 7; ++kh) {
        int ty = oy + 2 - kh;
        if (ty < 0 || (ty % 3)) continue;
        int iy = ty / 3;
        if (iy >= GW) continue;
#pragma unroll
        for (int kw = 0; kw < 7; ++kw) {
            int tx = ox + 2 - kw;
            if (tx < 0 || (tx % 3)) continue;
            int ix = tx / 3;
            if (ix >= GW) continue;
            const float* xp = x + (((size_t)(b * GW + iy)) * GW + ix) * 32;
            const float* wp = w + kh * 7 + kw;
#pragma unroll
            for (int c = 0; c < 32; ++c) acc += xp[c] * wp[c * 49];
        }
    }
    out[idx] = 1.f / (1.f + __expf(-acc));
}

// ---------------------------------------------------------------------------
extern "C" void kernel_launch(void* const* d_in, const int* in_sizes, int n_in,
                              void* d_out, int out_size, void* d_ws, size_t ws_size,
                              hipStream_t stream) {
    const float* features = (const float*)d_in[0];
    const int*   src      = (const int*)d_in[1];
    const int*   dst      = (const int*)d_in[2];
    const int*   ety      = (const int*)d_in[3];
    const float* norm     = (const float*)d_in[4];
    const float* bases[3] = {(const float*)d_in[5],  (const float*)d_in[9],  (const float*)d_in[13]};
    const float* comp[3]  = {(const float*)d_in[6],  (const float*)d_in[10], (const float*)d_in[14]};
    const float* loopw[3] = {(const float*)d_in[7],  (const float*)d_in[11], (const float*)d_in[15]};
    const float* bias[3]  = {(const float*)d_in[8],  (const float*)d_in[12], (const float*)d_in[16]};
    const float* conv_w   = (const float*)d_in[17];
    const float* conv_b   = (const float*)d_in[18];

    // workspace carving
    char* ws = (char*)d_ws;
    size_t off = 0;
    auto carve = [&](size_t bytes) {
        void* p = ws + off;
        off = (off + bytes + 255) & ~(size_t)255;
        return p;
    };
    float*  proj   = (float*)carve((size_t)NNODES * 512 * 4);   // worst-case 8*64 cols
    float*  agg    = (float*)carve((size_t)NNODES * 64 * 4);
    float*  h1     = (float*)carve((size_t)NNODES * 64 * 4);
    float*  h2     = (float*)carve((size_t)NNODES * 64 * 4);
    float*  logits = (float*)carve((size_t)NNODES * 32 * 4);
    __bf16* bpack  = (__bf16*)carve(80 * 1024);

    const int GEMM_BLOCKS = NNODES / (16 * 8);   // 900

    // ---- layer 0: K=32, DOUT=64 ----
    {
        constexpr int K = 32, DOUT = 64;
        constexpr int TOTAL = (K / 32) * (((RREL + 1) * DOUT) / 16) * 512;
        pack_w_kernel<K, DOUT><<<(TOTAL + 255) / 256, 256, 0, stream>>>(
            bases[0], comp[0], loopw[0], bpack);
        gemm_wmma_kernel<K, DOUT><<<GEMM_BLOCKS, 256, 0, stream>>>(
            features, bpack, bias[0], proj, agg);
        scatter_kernel<DOUT><<<(NEDGES * (DOUT / 4) + 255) / 256, 256, 0, stream>>>(
            proj, src, dst, ety, norm, agg);
        act_kernel<<<(NNODES * DOUT + 255) / 256, 256, 0, stream>>>(
            agg, h1, NNODES * DOUT, 0);
    }
    // ---- layer 1: K=64, DOUT=64 ----
    {
        constexpr int K = 64, DOUT = 64;
        constexpr int TOTAL = (K / 32) * (((RREL + 1) * DOUT) / 16) * 512;
        pack_w_kernel<K, DOUT><<<(TOTAL + 255) / 256, 256, 0, stream>>>(
            bases[1], comp[1], loopw[1], bpack);
        gemm_wmma_kernel<K, DOUT><<<GEMM_BLOCKS, 256, 0, stream>>>(
            h1, bpack, bias[1], proj, agg);
        scatter_kernel<DOUT><<<(NEDGES * (DOUT / 4) + 255) / 256, 256, 0, stream>>>(
            proj, src, dst, ety, norm, agg);
        act_kernel<<<(NNODES * DOUT + 255) / 256, 256, 0, stream>>>(
            agg, h2, NNODES * DOUT, 0);
    }
    // ---- layer 2: K=64, DOUT=32, sigmoid ----
    {
        constexpr int K = 64, DOUT = 32;
        constexpr int TOTAL = (K / 32) * (((RREL + 1) * DOUT) / 16) * 512;
        pack_w_kernel<K, DOUT><<<(TOTAL + 255) / 256, 256, 0, stream>>>(
            bases[2], comp[2], loopw[2], bpack);
        gemm_wmma_kernel<K, DOUT><<<GEMM_BLOCKS, 256, 0, stream>>>(
            h2, bpack, bias[2], proj, agg);
        scatter_kernel<DOUT><<<(NEDGES * (DOUT / 4) + 255) / 256, 256, 0, stream>>>(
            proj, src, dst, ety, norm, agg);
        act_kernel<<<(NNODES * DOUT + 255) / 256, 256, 0, stream>>>(
            agg, logits, NNODES * DOUT, 1);
    }
    // ---- decoder ----
    convt_sigmoid_kernel<<<(BGRAPH * IW * IW + 255) / 256, 256, 0, stream>>>(
        logits, conv_w, conv_b, (float*)d_out);
}
